// SinkhornDistillationLoss_17970143167373
// MI455X (gfx1250) — matrix operand order repaired and checked by reading, not compiled, not run on hardware
//
#include <hip/hip_runtime.h>
#include <stdint.h>

#define B_SZ 64
#define K_SZ 1000
#define KP   1024                 // padded K for WMMA tiling
#define USTR 1032                 // LDS row stride (bank-conflict-free: 2064B/4 mod 64 = 4)
#define TEMP_INV 0.25f
#define INV_EPS  20.0f            // 1/0.05
#define LOG_CLAMP -18.420680743952367f   // log(1e-8)
#define THRESH 1e-3f
#define MAX_ITER 50

typedef __attribute__((ext_vector_type(8)))  float  v8f;
typedef __attribute__((ext_vector_type(16))) __bf16 v16bf;

__device__ __forceinline__ unsigned short f2bf(float f) {
  unsigned u = __float_as_uint(f);
  unsigned r = u + 0x7FFFu + ((u >> 16) & 1u);   // round-to-nearest-even
  return (unsigned short)(r >> 16);
}

// ---------------------------------------------------------------------------
// Kernel 1: Gibbs kernel exp(-C/eps) in bf16, pre-swizzled into the WMMA
// B-fragment layout in this file's contiguous k order (A is loaded to match):
//   lane<16 : K = e,      N = lane
//   lane>=16: K = e + 16, N = lane - 16
// KB1: B = Kmat^T (contraction index = j) for the f half-step
// KB2: B = Kmat   (contraction index = i) for the g half-step
// Element index t = ((tile*32 + kk)*32 + lane)*16 + e  → one 32B load/lane.
// ---------------------------------------------------------------------------
__global__ __launch_bounds__(256) void gibbs_swizzle_kernel(
    const float* __restrict__ C,
    unsigned short* __restrict__ KB1,
    unsigned short* __restrict__ KB2)
{
  unsigned t = blockIdx.x * 256u + threadIdx.x;    // 0 .. 2^20-1
  int e    = t & 15;
  int lane = (t >> 4) & 31;
  int kk   = (t >> 9) & 31;
  int tile = t >> 14;                              // 0..63
  int kdim = kk * 32 + ((lane >> 4) ? 16 : 0) + e; // contraction index 0..1023
  int n    = tile * 16 + (lane & 15);              // output (N) index  0..1023
  float v1 = 0.f, v2 = 0.f;
  if (kdim < K_SZ && n < K_SZ) {
    v1 = expf(-C[(size_t)n * K_SZ + kdim] * INV_EPS);  // KB1[j=kdim][i=n] = Kmat[i,j]
    v2 = expf(-C[(size_t)kdim * K_SZ + n] * INV_EPS);  // KB2[i=kdim][j=n] = Kmat[i,j]
  }
  KB1[t] = f2bf(v1);
  KB2[t] = f2bf(v2);
}

// ---------------------------------------------------------------------------
// Row logsumexp helper: 16 threads per row (l = tid&15), 16 rows per block.
// Returns (rowmax, log(sum exp(x*scale - rowmax))).
// ---------------------------------------------------------------------------
__device__ __forceinline__ float2 row_logsumexp(const float* __restrict__ row,
                                                float scale, float* red,
                                                int tid, int l, int r)
{
  float pm = -3.4e38f;
  for (int j = l; j < K_SZ; j += 16) pm = fmaxf(pm, row[j] * scale);
  red[tid] = pm; __syncthreads();
  if (l == 0) { float m = red[tid];
    for (int k = 1; k < 16; ++k) m = fmaxf(m, red[tid + k]);
    red[tid] = m; }
  __syncthreads();
  float m = red[r * 16];
  __syncthreads();
  float ps = 0.f;
  for (int j = l; j < K_SZ; j += 16) ps += expf(row[j] * scale - m);
  red[tid] = ps; __syncthreads();
  if (l == 0) { float s = red[tid];
    for (int k = 1; k < 16; ++k) s += red[tid + k];
    red[tid] = s; }
  __syncthreads();
  float lS = logf(red[r * 16]);
  __syncthreads();
  return make_float2(m, lS);
}

// ---------------------------------------------------------------------------
// Kernel 2: persistent Sinkhorn. 4 blocks x 256 threads (8 waves).
// Each block owns 16 batch rows (WMMA M=16). f,g live in LDS (CDNA5 320KB).
// Each half-step: row-max + exp into bf16 LDS tile, then 64 i-tiles x 32
// v_wmma_f32_16x16x32_bf16 against the L2-resident swizzled Gibbs kernel.
// ---------------------------------------------------------------------------
__global__ __launch_bounds__(256, 1) void sinkhorn_wmma_kernel(
    const float* __restrict__ student,
    const float* __restrict__ teacher,
    const int*   __restrict__ labels,
    const unsigned short* __restrict__ KB1,
    const unsigned short* __restrict__ KB2,
    float* __restrict__ la, float* __restrict__ lb,
    float* __restrict__ U,  float* __restrict__ V,
    float* __restrict__ cePart)
{
  __shared__ float fS[16][KP];                 // 64 KB
  __shared__ float gS[16][KP];                 // 64 KB
  __shared__ unsigned short uS[16][USTR];      // 33 KB  (bf16 A-operand tile)
  __shared__ float red[256];
  __shared__ float wmaxS[16];
  __shared__ unsigned errBits;
  __shared__ int doneS;

  const int tid = threadIdx.x;
  const int r = tid >> 4, l = tid & 15;        // row / lane-in-row for reductions
  const int b0 = blockIdx.x * 16;
  const int row = b0 + r;
  const int wv = tid >> 5, lane = tid & 31;    // wave id / lane for WMMA
  const int halfL = lane >> 4, nl = lane & 15, koff = halfL * 8;

  // ---- phase 0: log p_T -> la, log p_S -> lb (T=4), CE (T=1) ----
  {
    const float* trow = teacher + (size_t)row * K_SZ;
    float2 ml = row_logsumexp(trow, TEMP_INV, red, tid, l, r);
    for (int j = l; j < K_SZ; j += 16)
      la[(size_t)row * KP + j] = fmaxf(trow[j] * TEMP_INV - ml.x - ml.y, LOG_CLAMP);
  }
  {
    const float* srow = student + (size_t)row * K_SZ;
    float2 ml = row_logsumexp(srow, TEMP_INV, red, tid, l, r);
    for (int j = l; j < K_SZ; j += 16)
      lb[(size_t)row * KP + j] = fmaxf(srow[j] * TEMP_INV - ml.x - ml.y, LOG_CLAMP);
    float2 m1 = row_logsumexp(srow, 1.0f, red, tid, l, r);
    if (l == 0) cePart[row] = (m1.x + m1.y) - srow[labels[row]];
  }
  for (int k = tid; k < 16 * KP; k += 256) { (&fS[0][0])[k] = 0.f; (&gS[0][0])[k] = 0.f; }
  if (tid == 0) doneS = 0;
  __syncthreads();

  for (int iter = 0; iter < MAX_ITER; ++iter) {
    if (doneS) break;                          // uniform (barrier-protected)
    if (tid == 0) errBits = 0u;

    // ================= f half-step: u = exp(g + lb - wmax) =================
    {
      const float* Lrow = lb + (size_t)row * KP;
      float pm = -3.4e38f;
      for (int j = l; j < K_SZ; j += 16) pm = fmaxf(pm, gS[r][j] + Lrow[j]);
      red[tid] = pm; __syncthreads();
      if (l == 0) { float m = red[tid];
        for (int k = 1; k < 16; ++k) m = fmaxf(m, red[tid + k]);
        wmaxS[r] = m; }
      __syncthreads();
      float m = wmaxS[r];
      for (int j = l; j < K_SZ; j += 16) uS[r][j] = f2bf(expf(gS[r][j] + Lrow[j] - m));
      for (int j = K_SZ + l; j < USTR; j += 16) uS[r][j] = 0;
      __syncthreads();
    }
    // f_new = -(wmax + log(u @ Kmat^T)) via WMMA
    for (int it = wv; it < 64; it += 8) {
      v8f acc = {0.f,0.f,0.f,0.f,0.f,0.f,0.f,0.f};
      const unsigned short* kb = KB1 + ((size_t)it << 14) + ((size_t)lane << 4);
      for (int kk = 0; kk < 32; ++kk) {
        union { v16bf v; unsigned short s[16]; } A;
        #pragma unroll
        for (int e = 0; e < 8; ++e) {          // A 16x32: K = 16*half + e (matches B swizzle)
          A.s[e]     = uS[nl][kk * 32 + halfL * 16 + e];
          A.s[e + 8] = uS[nl][kk * 32 + halfL * 16 + 8 + e];
        }
        union { v16bf v; uint4 q[2]; } Bf;
        const uint4* p = (const uint4*)(kb + ((size_t)kk << 9));
        Bf.q[0] = p[0]; Bf.q[1] = p[1];
        acc = __builtin_amdgcn_wmma_f32_16x16x32_bf16(false, A.v, false, Bf.v,
                                                      (short)0, acc, false, false);
      }
      const int i0 = it * 16;
      #pragma unroll
      for (int rr = 0; rr < 8; ++rr) {         // ISA f32 C/D 16x16 layout
        int brow = rr + halfL * 8;
        int i = i0 + nl;
        float fn = -(wmaxS[brow] + logf(fmaxf(acc[rr], 1e-37f)));
        if (i < K_SZ) {
          float d = fabsf(fn - fS[brow][i]);
          atomicMax(&errBits, __float_as_uint(d));   // |d|>=0: uint order == float order
        }
        fS[brow][i] = fn;
      }
    }
    __syncthreads();

    // ================= g half-step: v = exp(f_new + la - fmax) =============
    {
      const float* Lrow = la + (size_t)row * KP;
      float pm = -3.4e38f;
      for (int j = l; j < K_SZ; j += 16) pm = fmaxf(pm, fS[r][j] + Lrow[j]);
      red[tid] = pm; __syncthreads();
      if (l == 0) { float m = red[tid];
        for (int k = 1; k < 16; ++k) m = fmaxf(m, red[tid + k]);
        wmaxS[r] = m; }
      __syncthreads();
      float m = wmaxS[r];
      for (int j = l; j < K_SZ; j += 16) uS[r][j] = f2bf(expf(fS[r][j] + Lrow[j] - m));
      for (int j = K_SZ + l; j < USTR; j += 16) uS[r][j] = 0;
      __syncthreads();
    }
    for (int it = wv; it < 64; it += 8) {
      v8f acc = {0.f,0.f,0.f,0.f,0.f,0.f,0.f,0.f};
      const unsigned short* kb = KB2 + ((size_t)it << 14) + ((size_t)lane << 4);
      for (int kk = 0; kk < 32; ++kk) {
        union { v16bf v; unsigned short s[16]; } A;
        #pragma unroll
        for (int e = 0; e < 8; ++e) {
          A.s[e]     = uS[nl][kk * 32 + halfL * 16 + e];
          A.s[e + 8] = uS[nl][kk * 32 + halfL * 16 + 8 + e];
        }
        union { v16bf v; uint4 q[2]; } Bf;
        const uint4* p = (const uint4*)(kb + ((size_t)kk << 9));
        Bf.q[0] = p[0]; Bf.q[1] = p[1];
        acc = __builtin_amdgcn_wmma_f32_16x16x32_bf16(false, A.v, false, Bf.v,
                                                      (short)0, acc, false, false);
      }
      const int j0 = it * 16;
      #pragma unroll
      for (int rr = 0; rr < 8; ++rr) {
        int brow = rr + halfL * 8;
        gS[brow][j0 + nl] = -(wmaxS[brow] + logf(fmaxf(acc[rr], 1e-37f)));
      }
    }
    __syncthreads();
    if (tid == 0 && __uint_as_float(errBits) < THRESH) doneS = 1;
    __syncthreads();
  }

  __syncthreads();
  // ---- export scaling vectors U = exp(f+la), V = exp(g+lb) ----
  {
    const float* Larow = la + (size_t)row * KP;
    const float* Lbrow = lb + (size_t)row * KP;
    for (int j = l; j < K_SZ; j += 16) {
      U[(size_t)row * KP + j] = expf(fS[r][j] + Larow[j]);
      V[(size_t)row * KP + j] = expf(gS[r][j] + Lbrow[j]);
    }
  }
}

// ---------------------------------------------------------------------------
// Kernel 3: pi[b,i,j] = U[b,i] * exp(-C[i,j]/eps) * V[b,j]  (256 MB stream)
// + per-block partial of sum_b sum_ij C_ij * pi.
// grid = (4 j-chunks, 1000 i-rows), block = 256.
// ---------------------------------------------------------------------------
__global__ __launch_bounds__(256) void plan_cost_kernel(
    const float* __restrict__ C,
    const float* __restrict__ U,
    const float* __restrict__ V,
    float* __restrict__ out,          // d_out: [0..2] losses, [3..] plan
    float* __restrict__ costPart)
{
  __shared__ float Ui[B_SZ];
  __shared__ float rs[256];
  const int tid = threadIdx.x;
  const int i = blockIdx.y;
  const int j = blockIdx.x * 256 + tid;
  if (tid < B_SZ) Ui[tid] = U[(size_t)tid * KP + i];
  __syncthreads();

  float csum = 0.f;
  if (j < K_SZ) {
    float c  = C[(size_t)i * K_SZ + j];
    float kf = expf(-c * INV_EPS);
    float* o = out + 3 + (size_t)i * K_SZ + j;
    #pragma unroll 4
    for (int b = 0; b < B_SZ; ++b) {
      float p = Ui[b] * kf * V[(size_t)b * KP + j];
      o[(size_t)b * K_SZ * K_SZ] = p;
      csum += c * p;
    }
  }
  rs[tid] = csum; __syncthreads();
  for (int st = 128; st > 0; st >>= 1) {
    if (tid < st) rs[tid] += rs[tid + st];
    __syncthreads();
  }
  if (tid == 0) costPart[blockIdx.y * 4 + blockIdx.x] = rs[0];
}

// ---------------------------------------------------------------------------
// Kernel 4: deterministic final reduction -> total / ot / ce.
// ---------------------------------------------------------------------------
__global__ __launch_bounds__(256) void finalize_kernel(
    const float* __restrict__ cePart,
    const float* __restrict__ costPart,
    float* __restrict__ out)
{
  __shared__ float rs[256];
  const int tid = threadIdx.x;
  float s = 0.f;
  for (int k = tid; k < 4 * K_SZ; k += 256) s += costPart[k];
  rs[tid] = s; __syncthreads();
  for (int st = 128; st > 0; st >>= 1) {
    if (tid < st) rs[tid] += rs[tid + st];
    __syncthreads();
  }
  if (tid == 0) {
    float ce = 0.f;
    for (int k = 0; k < B_SZ; ++k) ce += cePart[k];
    ce *= (1.0f / B_SZ);
    float ot = rs[0] * (1.0f / B_SZ);
    out[0] = ce + 0.5f * ot;
    out[1] = ot;
    out[2] = ce;
  }
}

// ---------------------------------------------------------------------------
extern "C" void kernel_launch(void* const* d_in, const int* in_sizes, int n_in,
                              void* d_out, int out_size, void* d_ws, size_t ws_size,
                              hipStream_t stream)
{
  (void)in_sizes; (void)n_in; (void)out_size; (void)ws_size;
  const float* student = (const float*)d_in[0];
  const float* teacher = (const float*)d_in[1];
  const int*   labels  = (const int*)d_in[2];
  const float* C       = (const float*)d_in[3];
  float* out = (float*)d_out;

  char* w = (char*)d_ws;
  const size_t ROWB = (size_t)B_SZ * KP * sizeof(float);   // 256 KB
  float* la = (float*)(w);
  float* lb = (float*)(w + ROWB);
  float* U  = (float*)(w + 2 * ROWB);
  float* V  = (float*)(w + 3 * ROWB);
  unsigned short* KB1 = (unsigned short*)(w + 4 * ROWB);                       // 2 MB
  unsigned short* KB2 = (unsigned short*)(w + 4 * ROWB + (size_t)(1u << 21)); // 2 MB
  float* cePart   = (float*)(w + 4 * ROWB + (size_t)(2u << 21));
  float* costPart = cePart + B_SZ;

  gibbs_swizzle_kernel<<<4096, 256, 0, stream>>>(C, KB1, KB2);
  sinkhorn_wmma_kernel<<<4, 256, 0, stream>>>(student, teacher, labels,
                                              KB1, KB2, la, lb, U, V, cePart);
  plan_cost_kernel<<<dim3(4, K_SZ), 256, 0, stream>>>(C, U, V, out, costPart);
  finalize_kernel<<<1, 256, 0, stream>>>(cePart, costPart, out);
}
